// OurModel_88141318848640
// MI455X (gfx1250) — compile-verified, matrix operand order checked by reading, not run on hardware
//
#include <hip/hip_runtime.h>
#include <hip/hip_bf16.h>

typedef __attribute__((ext_vector_type(16))) __bf16 v16bf;
typedef __attribute__((ext_vector_type(8)))  float  v8f;
typedef unsigned short u16;
typedef unsigned int   u32;

// ---------------------------------------------------------------- helpers
__device__ __forceinline__ u16 f2bf(float f) {
    u32 x = __float_as_uint(f);
    u32 r = x + 0x7FFFu + ((x >> 16) & 1u);   // round-to-nearest-even
    return (u16)(r >> 16);
}

union Frag {
    v16bf v;
    uint4 q[2];
};

// ---------------------------------------------------------------- elementwise
__global__ void __launch_bounds__(256)
k_f32_to_bf16(const float* __restrict__ in, u16* __restrict__ out, size_t n) {
    size_t i = (size_t)blockIdx.x * blockDim.x + threadIdx.x;
    if (i < n) out[i] = f2bf(in[i]);
}

// W (K x M row-major, fp32) -> Wt (M x Kpad row-major, bf16, zero-padded K)
__global__ void __launch_bounds__(256)
k_transpose_bf16(const float* __restrict__ W, u16* __restrict__ Wt,
                 int K, int M, int Kpad, size_t n) {
    size_t i = (size_t)blockIdx.x * blockDim.x + threadIdx.x;
    if (i >= n) return;
    int m = (int)(i / (size_t)Kpad);
    int k = (int)(i % (size_t)Kpad);
    Wt[i] = (k < K) ? f2bf(W[(size_t)k * M + m]) : (u16)0;
}

__global__ void __launch_bounds__(256)
k_fill_zero(float* __restrict__ p, size_t n) {
    size_t i = (size_t)blockIdx.x * blockDim.x + threadIdx.x;
    if (i < n) p[i] = 0.0f;
}

// S (N x D fp32) + bias -> tanh -> bf16 (D is power of two here: mask = D-1)
__global__ void __launch_bounds__(256)
k_bias_tanh_bf16(const float* __restrict__ S, const float* __restrict__ bias,
                 u16* __restrict__ out, int dmask, size_t n) {
    size_t i = (size_t)blockIdx.x * blockDim.x + threadIdx.x;
    if (i >= n) return;
    int col = (int)(i & (size_t)dmask);
    out[i] = f2bf(tanhf(S[i] + bias[col]));
}

// S (N x 128 fp32) + bias -> dropout(p=0.5, deterministic hash) -> bf16
__global__ void __launch_bounds__(256)
k_bias_dropout_bf16(const float* __restrict__ S, const float* __restrict__ bias,
                    u16* __restrict__ out, size_t n) {
    size_t i = (size_t)blockIdx.x * blockDim.x + threadIdx.x;
    if (i >= n) return;
    int col = (int)(i & 127u);
    u32 h = (u32)i * 2654435761u;
    h ^= h >> 16; h *= 2246822519u; h ^= h >> 13;
    float v = S[i] + bias[col];
    out[i] = f2bf((h & 1u) ? v * 2.0f : 0.0f);
}

// fp32 (N x D) -> bf16 (N x Dpad) with zero fill in [D, Dpad)
__global__ void __launch_bounds__(256)
k_pad_bf16(const float* __restrict__ in, u16* __restrict__ out,
           int D, int Dpad, size_t n) {
    size_t i = (size_t)blockIdx.x * blockDim.x + threadIdx.x;
    if (i >= n) return;
    int row = (int)(i / (size_t)Dpad);
    int col = (int)(i % (size_t)Dpad);
    out[i] = (col < D) ? f2bf(in[(size_t)row * D + col]) : (u16)0;
}

// ---------------------------------------------------------------- SpMM scatter
// one wave per edge; lanes stride the feature dim in float4 chunks
__global__ void __launch_bounds__(256)
k_spmm(const int* __restrict__ rows, const int* __restrict__ cols,
       const float* __restrict__ vals, const float* __restrict__ H,
       float* __restrict__ Out, int D, int E) {
    int wave = blockIdx.x * (blockDim.x >> 5) + (threadIdx.x >> 5);
    int lane = threadIdx.x & 31;
    if (wave >= E) return;
    int r = rows[wave];
    int c = cols[wave];
    float v = vals[wave];
    const float4* __restrict__ src = (const float4*)(H + (size_t)c * D);
    float* __restrict__ dst = Out + (size_t)r * D;
    int nd4 = D >> 2;
    for (int d4 = lane; d4 < nd4; d4 += 32) {
        float4 h4 = src[d4];
        atomicAdd(dst + d4 * 4 + 0, v * h4.x);
        atomicAdd(dst + d4 * 4 + 1, v * h4.y);
        atomicAdd(dst + d4 * 4 + 2, v * h4.z);
        atomicAdd(dst + d4 * 4 + 3, v * h4.w);
    }
}

// ---------------------------------------------------------------- WMMA GEMM
// C[nrows x M] = A[nrows x K](bf16, lda = padded K) * Bt[M x K](bf16)^T
// Block: 8 waves = 128 rows x 64 cols. B tile (64 cols x 64 K bf16) staged in
// LDS, double-buffered (18 KB), cooperatively loaded (2 uint4/thread/stage,
// staged unconditionally with clamped offset -> branch-free loop body).
// 144 B LDS row pitch -> b128 fragment reads are bank-conflict free.
// Per barrier: 2 A frags (global), 16 ds_load_b128 into distinct regs,
// 8 back-to-back WMMAs. K must be a multiple of 64 (pre-padded);
// nrows a multiple of 16. Row/col guards hoisted via pointer clamping.
#define KSTEP  64
#define LPITCH 72    // u16 per LDS row: 64 data + 8 pad (144 B pitch)
__global__ void __launch_bounds__(256)
k_gemm_wmma(const u16* __restrict__ A, int lda,
            const u16* __restrict__ Bt,
            const float* __restrict__ bias,
            float* __restrict__ C, int ldc,
            int nrows, int K, int M, int act) {
    __shared__ u16 lds[2 * 64 * LPITCH];   // 18,432 B

    int tid    = threadIdx.x;
    int waveid = tid >> 5;
    int lane   = tid & 31;
    int row0 = blockIdx.y * 128 + waveid * 16;
    int col0 = blockIdx.x * 64;
    int r  = lane & 15;
    int kb = (lane >> 4) << 3;               // 0 or 8

    // --- B staging (per thread): 32 B (2 uint4) of the 64x64 tile per stage
    int sCol = tid >> 2;                     // 0..63 (local column)
    int sOff = (tid & 3) * 16;               // u16 offset in K: 0,16,32,48
    int sColG = col0 + sCol; if (sColG >= M) sColG = M - 1;   // clamp: masked at store
    const u16* sPtr = Bt + (size_t)sColG * K + sOff;
    u16* sDst = &lds[sCol * LPITCH + sOff];

    // --- A pointer (row clamped for tail block; stores guarded)
    int ra = row0 + r; if (ra >= nrows) ra = nrows - 1;
    const u16* aptr = A + (size_t)ra * lda + kb;

    v8f acc[4];
    acc[0] = (v8f)(0.0f); acc[1] = (v8f)(0.0f);
    acc[2] = (v8f)(0.0f); acc[3] = (v8f)(0.0f);

    // prologue: stage k=0 into buffer 0
    {
        uint4 q0 = *(const uint4*)(sPtr);
        uint4 q1 = *(const uint4*)(sPtr + 8);
        *(uint4*)(sDst)     = q0;
        *(uint4*)(sDst + 8) = q1;
    }

    int p = 0;
    for (int k0 = 0; k0 < K; k0 += KSTEP) {
        __syncthreads();                                  // staged tile visible

        // stage next tile unconditionally (clamped; redundant on last iter)
        int kn = (k0 + KSTEP < K) ? (k0 + KSTEP) : 0;
        uint4 n0 = *(const uint4*)(sPtr + kn);
        uint4 n1 = *(const uint4*)(sPtr + kn + 8);
        u16* d = sDst + (p ^ 1) * (64 * LPITCH);
        *(uint4*)(d)     = n0;
        *(uint4*)(d + 8) = n1;

        // A fragments for this 64-K step
        Frag a0, a1;
        a0.q[0] = *(const uint4*)(aptr + k0);
        a0.q[1] = *(const uint4*)(aptr + k0 + 16);
        a1.q[0] = *(const uint4*)(aptr + k0 + 32);
        a1.q[1] = *(const uint4*)(aptr + k0 + 48);
        __builtin_prefetch(aptr + k0 + 128, 0, 1);        // global_prefetch_b8

        // batch all 16 LDS fragment loads into distinct registers,
        // then issue 8 WMMAs back-to-back (single DS wait)
        const u16* lb = &lds[p * (64 * LPITCH) + kb];
        Frag b0[4], b1[4];
#pragma unroll
        for (int t = 0; t < 4; ++t) {
            const u16* bp = lb + (t * 16 + r) * LPITCH;
            b0[t].q[0] = *(const uint4*)(bp);
            b0[t].q[1] = *(const uint4*)(bp + 16);
            b1[t].q[0] = *(const uint4*)(bp + 32);
            b1[t].q[1] = *(const uint4*)(bp + 48);
        }
#pragma unroll
        for (int t = 0; t < 4; ++t)
            acc[t] = __builtin_amdgcn_wmma_f32_16x16x32_bf16(
                false, a0.v, false, b0[t].v, (short)0, acc[t], false, false);
#pragma unroll
        for (int t = 0; t < 4; ++t)
            acc[t] = __builtin_amdgcn_wmma_f32_16x16x32_bf16(
                false, a1.v, false, b1[t].v, (short)0, acc[t], false, false);

        p ^= 1;
    }

    if (row0 >= nrows) return;                            // tail waves: after last barrier
    int rbase = (lane < 16) ? 0 : 8;
#pragma unroll
    for (int t = 0; t < 4; ++t) {
        int col = col0 + t * 16 + (lane & 15);
        if (col >= M) continue;
        float bv = bias ? bias[col] : 0.0f;
#pragma unroll
        for (int i = 0; i < 8; ++i) {
            int row = row0 + i + rbase;
            float v = acc[t][i] + bv;
            if (act == 1) v = fmaxf(v, 0.0f);
            C[(size_t)row * ldc + col] = v;
        }
    }
}

// ---------------------------------------------------------------- launch
static inline unsigned gsz(size_t n) { return (unsigned)((n + 255) / 256); }

extern "C" void kernel_launch(void* const* d_in, const int* in_sizes, int n_in,
                              void* d_out, int out_size, void* d_ws, size_t ws_size,
                              hipStream_t stream) {
    const int N = 100000;
    const int E = 3200000;

    const float* x   = (const float*)d_in[0];
    const float* ev  = (const float*)d_in[1];
    const float* W1  = (const float*)d_in[2];
    const float* b1  = (const float*)d_in[3];
    const float* W2  = (const float*)d_in[4];
    const float* b2  = (const float*)d_in[5];
    const float* W3  = (const float*)d_in[6];
    const float* b3  = (const float*)d_in[7];
    const float* fw1 = (const float*)d_in[8];
    const float* fb1 = (const float*)d_in[9];
    const float* fw2 = (const float*)d_in[10];
    const float* fb2 = (const float*)d_in[11];
    const float* fw3 = (const float*)d_in[12];
    const float* fb3 = (const float*)d_in[13];
    const int* erow  = (const int*)d_in[14];
    const int* ecol  = (const int*)d_in[15];
    float* outp = (float*)d_out;
    char* ws = (char*)d_ws;

    // workspace layout (all offsets 128B+ aligned; row strides keep 16B alignment)
    size_t o_xb   = 0;                                   // N*256   bf16
    size_t o_w1t  = o_xb   + (size_t)N * 256 * 2;        // 1024*256 bf16
    size_t o_w2t  = o_w1t  + 1024ull * 256 * 2;          // 512*1024 bf16
    size_t o_w3t  = o_w2t  + 512ull * 1024 * 2;          // 128*512  bf16
    size_t o_fw1t = o_w3t  + 128ull * 512 * 2;           // 152*128  bf16
    size_t o_fw2t = o_fw1t + 152ull * 128 * 2;           // 48*192   bf16
    size_t o_fw3t = o_fw2t + 48ull * 192 * 2;            // 1*64     bf16
    size_t o_g    = o_fw3t + 256;                        // N*1024 f32 (GEMM out)
    size_t o_s    = o_g    + (size_t)N * 1024 * 4;       // N*1024 f32 (SpMM out)
    size_t o_h    = o_s    + (size_t)N * 1024 * 4;       // N*1024 bf16 (activations)

    u16*   xb   = (u16*)(ws + o_xb);
    u16*   w1t  = (u16*)(ws + o_w1t);
    u16*   w2t  = (u16*)(ws + o_w2t);
    u16*   w3t  = (u16*)(ws + o_w3t);
    u16*   fw1t = (u16*)(ws + o_fw1t);
    u16*   fw2t = (u16*)(ws + o_fw2t);
    u16*   fw3t = (u16*)(ws + o_fw3t);
    float* gbuf = (float*)(ws + o_g);
    float* sbuf = (float*)(ws + o_s);
    u16*   hb   = (u16*)(ws + o_h);

    dim3 blk(256);
    const unsigned gy = (unsigned)((N + 127) / 128);       // row blocks (8 waves x 16 rows)
    const unsigned spmm_grid = (unsigned)((E + 7) / 8);    // 8 waves/block, 1 edge/wave

    // ---- convert inputs / weights to bf16 (weights transposed, K padded to x64)
    k_f32_to_bf16<<<gsz((size_t)N * 256), blk, 0, stream>>>(x, xb, (size_t)N * 256);
    k_transpose_bf16<<<gsz(1024ull * 256), blk, 0, stream>>>(W1, w1t, 256, 1024, 256, 1024ull * 256);
    k_transpose_bf16<<<gsz(512ull * 1024), blk, 0, stream>>>(W2, w2t, 1024, 512, 1024, 512ull * 1024);
    k_transpose_bf16<<<gsz(128ull * 512), blk, 0, stream>>>(W3, w3t, 512, 128, 512, 128ull * 512);
    k_transpose_bf16<<<gsz(152ull * 128), blk, 0, stream>>>(fw1, fw1t, 128, 152, 128, 152ull * 128);
    k_transpose_bf16<<<gsz(48ull * 192), blk, 0, stream>>>(fw2, fw2t, 152, 48, 192, 48ull * 192);
    k_transpose_bf16<<<gsz(64ull), blk, 0, stream>>>(fw3, fw3t, 48, 1, 64, 64ull);

    // ---- layer 1: g = X @ W1 ; s = A g ; h = tanh(s + b1)
    k_gemm_wmma<<<dim3(1024 / 64, gy), blk, 0, stream>>>(xb, 256, w1t, nullptr, gbuf, 1024, N, 256, 1024, 0);
    k_fill_zero<<<gsz((size_t)N * 1024), blk, 0, stream>>>(sbuf, (size_t)N * 1024);
    k_spmm<<<spmm_grid, blk, 0, stream>>>(erow, ecol, ev, gbuf, sbuf, 1024, E);
    k_bias_tanh_bf16<<<gsz((size_t)N * 1024), blk, 0, stream>>>(sbuf, b1, hb, 1023, (size_t)N * 1024);

    // ---- layer 2
    k_gemm_wmma<<<dim3(512 / 64, gy), blk, 0, stream>>>(hb, 1024, w2t, nullptr, gbuf, 512, N, 1024, 512, 0);
    k_fill_zero<<<gsz((size_t)N * 512), blk, 0, stream>>>(sbuf, (size_t)N * 512);
    k_spmm<<<spmm_grid, blk, 0, stream>>>(erow, ecol, ev, gbuf, sbuf, 512, E);
    k_bias_tanh_bf16<<<gsz((size_t)N * 512), blk, 0, stream>>>(sbuf, b2, hb, 511, (size_t)N * 512);

    // ---- layer 3 (linear GC) + dropout
    k_gemm_wmma<<<dim3(128 / 64, gy), blk, 0, stream>>>(hb, 512, w3t, nullptr, gbuf, 128, N, 512, 128, 0);
    k_fill_zero<<<gsz((size_t)N * 128), blk, 0, stream>>>(sbuf, (size_t)N * 128);
    k_spmm<<<spmm_grid, blk, 0, stream>>>(erow, ecol, ev, gbuf, sbuf, 128, E);
    k_bias_dropout_bf16<<<gsz((size_t)N * 128), blk, 0, stream>>>(sbuf, b3, hb, (size_t)N * 128);

    // ---- dense head: relu(h fw1 + fb1), relu(. fw2 + fb2), . fw3 + fb3
    k_gemm_wmma<<<dim3((152 + 63) / 64, gy), blk, 0, stream>>>(hb, 128, fw1t, fb1, gbuf, 152, N, 128, 152, 1);
    k_pad_bf16<<<gsz((size_t)N * 192), blk, 0, stream>>>(gbuf, hb, 152, 192, (size_t)N * 192);

    k_gemm_wmma<<<dim3(1, gy), blk, 0, stream>>>(hb, 192, fw2t, fb2, gbuf, 48, N, 192, 48, 1);
    k_pad_bf16<<<gsz((size_t)N * 64), blk, 0, stream>>>(gbuf, hb, 48, 64, (size_t)N * 64);

    k_gemm_wmma<<<dim3(1, gy), blk, 0, stream>>>(hb, 64, fw3t, fb3, outp, 1, N, 64, 1, 0);
}
